// DomainAdversarialFusionMMD_41592463294941
// MI455X (gfx1250) — matrix operand order repaired and checked by reading, not compiled, not run on hardware
//
#include <hip/hip_runtime.h>
#include <hip/hip_bf16.h>
#include <stdint.h>

// ---- WMMA vector types (gfx1250, wave32) ----
typedef __attribute__((ext_vector_type(16))) _Float16 v16h;
typedef __attribute__((ext_vector_type(8)))  float    v8f;

#define NROWS   8192
#define DIM     384
#define KSTEPS  12            // 384 / 32
#define MTILES  512           // 8192 / 16
#define DW_PER_ARRAY (MTILES * KSTEPS * 256)   // dwords per packed operand array
#define BLOCKS_PER_MAT 4096   // 64 x 64 blocks of 128x128
#define NBLK    12288         // 3 * 4096

// f32 -> f16 (RNE) pair packed into one dword (lo = first K of the pair).
__device__ __forceinline__ uint32_t pack2_f16(float lo, float hi) {
    union { _Float16 h[2]; uint32_t u; } u;
    u.h[0] = (_Float16)lo;
    u.h[1] = (_Float16)hi;
    return u.u;
}

// Pack x and y (row-major f32 [8192][384]) into f16 WMMA fragment layouts.
// Array order in ws: [0]=xA, [1]=xB, [2]=yA, [3]=yB, each DW_PER_ARRAY dwords.
// Fragment storage: dword index = ((tile*12 + kk)*32 + lane)*8 + v  -> one
// 16x32 fragment is 1 KB contiguous; each lane owns 32 contiguous bytes.
//
// A layout (16x32 f16, ISA 7.12.2): lane L: row m = L&15, half = L>>4;
//   dword v: k = kk*32 + (v>>2)*16 + half*8 + (v&3)*2   (pair k, k+1)
// B layout (32x16 f16, row striped across lanes within a VGPR):
//   lane L: col n = L&15; dword v: k = kk*32 + (v>>2)*16 + (L>>4)*8 + (v&3)*2
__global__ void mmd_pack_kernel(const float* __restrict__ x,
                                const float* __restrict__ y,
                                uint32_t* __restrict__ ws) {
    uint32_t gid  = blockIdx.x * blockDim.x + threadIdx.x;  // exact grid
    uint32_t arr  = gid / DW_PER_ARRAY;        // 0..3
    uint32_t idx  = gid - arr * DW_PER_ARRAY;
    uint32_t v    = idx & 7u;
    uint32_t lane = (idx >> 3) & 31u;
    uint32_t t    = idx >> 8;
    uint32_t kk   = t % KSTEPS;
    uint32_t tile = t / KSTEPS;

    const float* src = (arr < 2) ? x : y;
    uint32_t row  = tile * 16 + (lane & 15u);
    uint32_t half = lane >> 4;
    uint32_t k;
    if ((arr & 1u) == 0u) { // A layout
        k = kk * 32 + ((v >> 2) << 4) + (half << 3) + ((v & 3u) << 1);
    } else {                // B layout
        k = kk * 32 + ((v >> 2) << 4) + (half << 3) + ((v & 3u) << 1);
    }
    const float* p = src + (size_t)row * DIM + k;
    ws[(size_t)arr * DW_PER_ARRAY + idx] = pack2_f16(p[0], p[1]);
}

// Exact fp32 row norms: sxy[0..8191] = ||x_i||^2, sxy[8192..] = ||y_i||^2.
__global__ void mmd_norm_kernel(const float* __restrict__ x,
                                const float* __restrict__ y,
                                float* __restrict__ sxy) {
    int bid = blockIdx.x;                 // 0..16383
    const float* src = (bid < NROWS) ? x : y;
    int row = bid & (NROWS - 1);
    float* dst = sxy + ((bid < NROWS) ? 0 : NROWS);
    __shared__ float red[128];
    float s = 0.f;
    for (int j = threadIdx.x; j < DIM; j += 128) {
        float t = src[(size_t)row * DIM + j];
        s += t * t;
    }
    red[threadIdx.x] = s;
    __syncthreads();
    for (int off = 64; off; off >>= 1) {
        if ((int)threadIdx.x < off) red[threadIdx.x] += red[threadIdx.x + off];
        __syncthreads();
    }
    if (threadIdx.x == 0) dst[row] = red[0];
}

// Async-copy one 1 KB B fragment (btile, kk) into LDS. Identity byte layout:
// lane L moves 32 bytes at (frag + L*32). Two b128 moves per lane.
// Tracked on ASYNCcnt (CDNA5 ISA).
__device__ __forceinline__ void stage_b_frag(const uint32_t* __restrict__ B,
                                             int btile, int kk, int lane,
                                             uint32_t* lds_frag) {
    const uint32_t* g = B + ((size_t)(btile * KSTEPS + kk) * 256 + lane * 8);
    uint32_t lds_addr = (uint32_t)(uintptr_t)(lds_frag + lane * 8); // addr[31:0] = LDS offset
    asm volatile(
        "global_load_async_to_lds_b128 %0, %1, off\n\t"
        "global_load_async_to_lds_b128 %0, %1, off offset:16"
        :: "v"(lds_addr), "v"(g) : "memory");
}

__device__ __forceinline__ void wait_async0() {
    asm volatile("s_wait_asynccnt 0" ::: "memory");
}

// One workgroup = 128x128 block of one Gram matrix. For xx/yy only the upper
// triangle of blocks is computed (bm<=bn); off-diagonal partials are doubled.
// 8 waves: wave w owns one 16-row strip x eight 16x16 tiles. B fragments are
// staged through LDS with double-buffered async copies (shared by all waves);
// per-tile LDS reads are software-pipelined 2 deep, and the whole K loop is
// fully unrolled so accumulators stay pinned (no cross-iteration reg copies).
__global__ void __launch_bounds__(256)
mmd_gemm_kernel(const uint32_t* __restrict__ ws,
                const float* __restrict__ sx,
                const float* __restrict__ sy,
                float* __restrict__ blockSums) {
    int mat = blockIdx.x >> 12;           // 0=xx, 1=yy, 2=xy
    int rem = blockIdx.x & 4095;
    int bm  = rem >> 6;
    int bn  = rem & 63;

    if (mat < 2 && bm > bn) {             // symmetric: skip lower triangle
        if (threadIdx.x == 0) blockSums[blockIdx.x] = 0.f;
        return;
    }

    const uint32_t* A = ws + (size_t)((mat == 1) ? 2 : 0) * DW_PER_ARRAY; // xA or yA
    const uint32_t* B = ws + (size_t)((mat == 0) ? 1 : 3) * DW_PER_ARRAY; // xB or yB
    const float* sa = (mat == 1) ? sy : sx;
    const float* sb = (mat == 0) ? sx : sy;

    int w    = threadIdx.x >> 5;
    int lane = threadIdx.x & 31;
    int tm   = bm * 8 + w;

    __shared__ __align__(32) uint32_t bstage[2][2048];  // 2 x 8 KB B panels

    v8f acc[8];
    v8f zero = {0.f, 0.f, 0.f, 0.f, 0.f, 0.f, 0.f, 0.f};
#pragma unroll
    for (int i = 0; i < 8; ++i) acc[i] = zero;

    // Prologue: stage k-step 0 (wave w stages fragment nt == w).
    stage_b_frag(B, bn * 8 + w, 0, lane, &bstage[0][w * 256]);
    wait_async0();
    __syncthreads();

#pragma unroll
    for (int kk = 0; kk < KSTEPS; ++kk) {
        int cur = kk & 1;
        if (kk + 1 < KSTEPS)   // overlap async copy of k+1 with WMMAs of k
            stage_b_frag(B, bn * 8 + w, kk + 1, lane, &bstage[1 - cur][w * 256]);

        const uint32_t* ap = A + ((size_t)(tm * KSTEPS + kk) * 256 + lane * 8);
        v16h af = *(const v16h*)ap;
        if (kk + 1 < KSTEPS)
            __builtin_prefetch(A + ((size_t)(tm * KSTEPS + kk + 1) * 256 + lane * 8), 0, 1);

        // 2-deep pipelined B reads from LDS: issue loads for nt+2 before
        // the WMMA of nt so ds latency overlaps matrix-core work.
        const uint32_t* bp = &bstage[cur][lane * 8];
        v16h b0 = *(const v16h*)(bp + 0 * 256);
        v16h b1 = *(const v16h*)(bp + 1 * 256);
#pragma unroll
        for (int nt = 0; nt < 8; ++nt) {
            v16h bnext = (nt + 2 < 8) ? *(const v16h*)(bp + (nt + 2) * 256) : b0;
            acc[nt] = __builtin_amdgcn_wmma_f32_16x16x32_f16(
                false, af, false, b0, (short)0, acc[nt], false, false);
            b0 = b1;
            b1 = bnext;
        }

        wait_async0();      // k+1 panel landed
        __syncthreads();    // all waves done reading panel `cur`
    }

    // C layout: lane L -> n = L&15; vgpr v -> m = v + (L>=16 ? 8 : 0)
    int half = lane >> 4;
    int nloc = lane & 15;
    v8f sv = *(const v8f*)(sa + tm * 16 + half * 8);  // 32B-aligned

    const float inv_dim2 = 1.0f / (384.0f * 384.0f);
    float s_local = 0.f;
#pragma unroll
    for (int nt = 0; nt < 8; ++nt) {
        float sbn = sb[(bn * 8 + nt) * 16 + nloc];
#pragma unroll
        for (int v = 0; v < 8; ++v) {
            float d2 = sv[v] + sbn - 2.0f * acc[nt][v];
            d2 = fmaxf(d2, 0.0f);
            float ki = d2 * inv_dim2;
            // 1/(2*sigma^2) for sigma = 0.1, 1.0, 10.0
            s_local += __expf(-50.0f  * ki)
                     + __expf(-0.5f   * ki)
                     + __expf(-0.005f * ki);
        }
    }

    // deterministic wave32 + LDS reduction
#pragma unroll
    for (int off = 16; off; off >>= 1) s_local += __shfl_xor(s_local, off, 32);
    __shared__ float wsum[8];
    if (lane == 0) wsum[w] = s_local;
    __syncthreads();
    if (threadIdx.x == 0) {
        float t = 0.f;
        for (int i = 0; i < 8; ++i) t += wsum[i];
        if (mat < 2 && bm < bn) t *= 2.0f;   // account for skipped transpose block
        blockSums[blockIdx.x] = t;
    }
}

// Deterministic final combine in f64: (Sxx + Syy - 2*Sxy) / N^2.
__global__ void mmd_final_reduce(const float* __restrict__ blockSums,
                                 float* __restrict__ out) {
    __shared__ double red[256];
    double s = 0.0;
    for (int i = threadIdx.x; i < NBLK; i += 256) {
        double wgt = (i >= 2 * BLOCKS_PER_MAT) ? -2.0 : 1.0;
        s += wgt * (double)blockSums[i];
    }
    red[threadIdx.x] = s;
    __syncthreads();
    for (int off = 128; off; off >>= 1) {
        if ((int)threadIdx.x < off) red[threadIdx.x] += red[threadIdx.x + off];
        __syncthreads();
    }
    if (threadIdx.x == 0)
        out[0] = (float)(red[0] / (double)((long long)NROWS * (long long)NROWS));
}

extern "C" void kernel_launch(void* const* d_in, const int* in_sizes, int n_in,
                              void* d_out, int out_size, void* d_ws, size_t ws_size,
                              hipStream_t stream) {
    const float* x = (const float*)d_in[0];
    const float* y = (const float*)d_in[1];
    float* out = (float*)d_out;

    uint32_t* wsu = (uint32_t*)d_ws;
    // ws layout (dwords): 4 packed operand arrays, then norms, then block sums
    float* sx        = (float*)(wsu + (size_t)4 * DW_PER_ARRAY);
    float* sy        = sx + NROWS;
    float* blockSums = sy + NROWS;
    // total ws need: 4*6291456 + (2*8192 + 12288)*4 = 25,280,512 bytes

    // 1) pack to f16 fragment layouts
    mmd_pack_kernel<<<(4 * DW_PER_ARRAY) / 256, 256, 0, stream>>>(x, y, wsu);
    // 2) exact fp32 row norms
    mmd_norm_kernel<<<2 * NROWS, 128, 0, stream>>>(x, y, sx);
    // 3) Gram reductions via f16 WMMA (xx/yy upper-triangular only)
    mmd_gemm_kernel<<<NBLK, 256, 0, stream>>>(wsu, sx, sy, blockSums);
    // 4) deterministic f64 combine
    mmd_final_reduce<<<1, 256, 0, stream>>>(blockSums, out);
}
